// CrossAttention_9216999817487
// MI455X (gfx1250) — compile-verified, matrix operand order checked by reading, not run on hardware
//
#include <hip/hip_runtime.h>
#include <hip/hip_bf16.h>

typedef __attribute__((ext_vector_type(16))) __bf16 v16bf;
typedef __attribute__((ext_vector_type(8)))  float  v8f;
typedef __attribute__((ext_vector_type(4)))  unsigned int v4u;
typedef __attribute__((ext_vector_type(8)))  int v8i_;
typedef __attribute__((ext_vector_type(4)))  int v4i_;

#define DIM   768
#define HEADS 12
#define HD    64
#define SEQ   2048
#define BATCH 2
#define MTOT  (BATCH * SEQ)   // 4096 rows of x/y flattened

// ---------- helpers ----------

__device__ __forceinline__ unsigned short f2bf(float f) {
    union { float f; unsigned int u; } c; c.f = f;
    unsigned int u = c.u;
    u += 0x7fffu + ((u >> 16) & 1u);   // round-to-nearest-even
    return (unsigned short)(u >> 16);
}

__device__ __forceinline__ unsigned pack2(float a, float b) {
    return (unsigned)f2bf(a) | ((unsigned)f2bf(b) << 16);
}

// generic pointer to LDS: low 32 bits are the LDS byte offset (LDS aperture)
__device__ __forceinline__ unsigned lds_off(const void* p) {
    return (unsigned)(uintptr_t)p;
}

// Tensor Data Mover: DMA a 2D tile (16-bit elements) from global to LDS.
// D# packing per CDNA5 ISA (group0 + group1; groups 2/3 zero
// for <=2D tensors). All operands wave-uniform -> SGPR descriptor groups.
// This toolchain exposes the 6-arg builtin (extra int32x8 descriptor group).
__device__ __forceinline__ void tdm_load_2d(unsigned ldsaddr, const void* gptr,
                                            unsigned tile_d0, unsigned tile_d1,
                                            unsigned tensor_d0, unsigned tensor_d1,
                                            unsigned stride_d0)
{
    unsigned long long ga = (unsigned long long)(uintptr_t)gptr;
    v4u g0;
    g0.x = 1u;                                              // count=1 (valid user D#)
    g0.y = ldsaddr;                                         // lds_addr [63:32]
    g0.z = (unsigned)(ga & 0xffffffffu);                    // global_addr[31:0]
    g0.w = (unsigned)((ga >> 32) & 0x01ffffffu) | (2u << 30); // addr[56:32] | type=2
    v8i_ g1;
    g1[0] = (int)(1u << 16);                                // data_size=1 -> 2 bytes
    g1[1] = (int)(tensor_d0 << 16);                         // tensor_dim0[15:0]
    g1[2] = (int)((tensor_d0 >> 16) & 0xffffu) | (int)(tensor_d1 << 16);
    g1[3] = (int)((tensor_d1 >> 16) & 0xffffu) | (int)(tile_d0 << 16);
    g1[4] = (int)(tile_d1 & 0xffffu);                       // tile_dim1 (tile_dim2=0)
    g1[5] = (int)stride_d0;                                 // tensor_dim0_stride[31:0]
    g1[6] = 0;                                              // stride hi, dim1_stride
    g1[7] = 0;
    v4i_ z4 = {0, 0, 0, 0};
    v8i_ z8 = {0, 0, 0, 0, 0, 0, 0, 0};
    __builtin_amdgcn_tensor_load_to_lds(g0, g1, z4, z4, z8, 0);
}

// Load a 16x32 bf16 fragment (A-layout; identical layout serves as the
// contiguous-along-K B operand of an NT gemm) from an LDS tile.
__device__ __forceinline__ v16bf load_frag(const unsigned short* base, int stride,
                                           int row0, int k0) {
    const int lane = threadIdx.x & 31;
    const unsigned short* p = base + (size_t)(row0 + (lane & 15)) * stride
                                   + k0 + ((lane >> 4) << 3);
    union { v16bf v; float4 f[2]; } u;
    u.f[0] = *(const float4*)(p);
    u.f[1] = *(const float4*)(p + 16);
    return u.v;
}

__device__ __forceinline__ v8f wmma_bf16(v16bf a, v16bf b, v8f c) {
    return __builtin_amdgcn_wmma_f32_16x16x32_bf16(false, a, false, b,
                                                   (short)0, c, false, false);
}

// ---------- stage 1: fused QKV projection (bf16 WMMA, f32 accum) ----------
// C[M=4096, N=768] = A[M,768] @ W[N,768]^T + bias  (NT gemm)
// grid.z: 0 -> Q from x, 1 -> K from y, 2 -> V from y
// Q,K written head-major [B,H,L,64]; V written TRANSPOSED [B,H,64,L] so the
// attention stage can TDM V chunks directly into [d][kv] LDS layout.

__global__ __launch_bounds__(256) void qkv_proj_kernel(
    const float* __restrict__ x, const float* __restrict__ y,
    const float* __restrict__ Wq, const float* __restrict__ bq,
    const float* __restrict__ Wk, const float* __restrict__ bk,
    const float* __restrict__ Wv, const float* __restrict__ bv,
    unsigned short* __restrict__ Qo, unsigned short* __restrict__ Ko,
    unsigned short* __restrict__ Vo)
{
    const int z = blockIdx.z;
    const float* A    = (z == 0) ? x  : y;
    const float* W    = (z == 0) ? Wq : (z == 1 ? Wk : Wv);
    const float* bias = (z == 0) ? bq : (z == 1 ? bk : bv);
    unsigned short* out = (z == 0) ? Qo : (z == 1 ? Ko : Vo);

    __shared__ unsigned short As[128][40];   // row stride 80B (16B aligned)
    __shared__ unsigned short Bs[128][40];

    const int tid  = threadIdx.x;
    const int wid  = tid >> 5;
    const int lane = tid & 31;
    const int wm = wid & 3;        // 4 wave-rows  -> 32 M each
    const int wn = wid >> 2;       // 2 wave-cols  -> 64 N each
    const int m0 = blockIdx.x * 128;
    const int n0 = blockIdx.y * 128;

    v8f acc[2][4] = {};

    const int lrow = tid >> 3;          // 0..31
    const int lcol = (tid & 7) * 4;     // float4 column within K-chunk

    for (int kt = 0; kt < DIM; kt += 32) {
        // speculative prefetch of next A chunk (dropped if OOB)
        __builtin_prefetch(A + (size_t)(m0 + lrow) * DIM + kt + 32, 0, 3);
        #pragma unroll
        for (int p = 0; p < 4; ++p) {
            int r = p * 32 + lrow;
            float4 v = *(const float4*)(A + (size_t)(m0 + r) * DIM + kt + lcol);
            *(uint2*)&As[r][lcol] = make_uint2(pack2(v.x, v.y), pack2(v.z, v.w));
            float4 w = *(const float4*)(W + (size_t)(n0 + r) * DIM + kt + lcol);
            *(uint2*)&Bs[r][lcol] = make_uint2(pack2(w.x, w.y), pack2(w.z, w.w));
        }
        __syncthreads();
        v16bf af[2], bfr[4];
        #pragma unroll
        for (int i = 0; i < 2; ++i) af[i]  = load_frag(&As[0][0], 40, wm * 32 + i * 16, 0);
        #pragma unroll
        for (int j = 0; j < 4; ++j) bfr[j] = load_frag(&Bs[0][0], 40, wn * 64 + j * 16, 0);
        #pragma unroll
        for (int i = 0; i < 2; ++i)
            #pragma unroll
            for (int j = 0; j < 4; ++j)
                acc[i][j] = wmma_bf16(af[i], bfr[j], acc[i][j]);
        __syncthreads();
    }

    const int grp = lane >> 4;
    const int cn  = lane & 15;
    #pragma unroll
    for (int i = 0; i < 2; ++i)
        #pragma unroll
        for (int j = 0; j < 4; ++j)
            #pragma unroll
            for (int r = 0; r < 8; ++r) {
                int gm = m0 + wm * 32 + i * 16 + r + 8 * grp;
                int gn = n0 + wn * 64 + j * 16 + cn;
                int b = gm >> 11, l = gm & 2047;
                int h = gn >> 6,  d = gn & 63;
                unsigned short val = f2bf(acc[i][j][r] + bias[gn]);
                if (z == 2)   // V: transposed [B,H,64,SEQ]
                    out[(((size_t)b * HEADS + h) * HD + d) * SEQ + l] = val;
                else          // Q,K: head-major [B,H,SEQ,64]
                    out[(((size_t)b * HEADS + h) * SEQ + l) * HD + d] = val;
            }
}

// ---------- stage 2: flash attention (TDM-fed LDS, online softmax) ----------
// grid: (SEQ/64, B*H), block: 128 threads (4 waves). Wave w owns query rows
// [l0+16w, l0+16w+16). KV in 32-row chunks, double-buffered via TENSORcnt.

__global__ __launch_bounds__(128) void attn_kernel(
    const unsigned short* __restrict__ Q, const unsigned short* __restrict__ K,
    const unsigned short* __restrict__ V, unsigned short* __restrict__ O)
{
    const int bh = blockIdx.y;
    const int l0 = blockIdx.x * 64;
    const size_t head_off = (size_t)bh * SEQ * HD;
    const unsigned short* Qh = Q + head_off;
    const unsigned short* Kh = K + head_off;
    const unsigned short* Vt = V + head_off;   // transposed head: [64][SEQ]

    __shared__ unsigned short Qs[64][64];       // query tile
    __shared__ unsigned short Ks[2][32][64];    // key chunk, double buffered
    __shared__ unsigned short Vs[2][64][32];    // value chunk [d][kv], dbl buf
    __shared__ unsigned short Ps[4][16][32];    // per-wave P relayout scratch

    const int tid  = threadIdx.x;
    const int wid  = tid >> 5;
    const int lane = tid & 31;

    if (wid == 0) {
        // Q tile + KV chunk 0 via Tensor Data Mover (contiguous / 2D tiles)
        tdm_load_2d(lds_off(&Qs[0][0]),    Qh + (size_t)l0 * HD,
                    64 * 64, 1, SEQ * HD, 1, 64 * 64);
        tdm_load_2d(lds_off(&Ks[0][0][0]), Kh,
                    32 * 64, 1, SEQ * HD, 1, 32 * 64);
        tdm_load_2d(lds_off(&Vs[0][0][0]), Vt,
                    32, 64, SEQ, HD, SEQ);
        __builtin_amdgcn_s_wait_tensorcnt(0);
    }
    __syncthreads();

    v16bf qa[2];
    qa[0] = load_frag(&Qs[0][0], 64, wid * 16, 0);
    qa[1] = load_frag(&Qs[0][0], 64, wid * 16, 32);

    const float scale = 0.125f;      // hd^-0.5 = 1/8
    v8f   o_acc[4] = {};
    float mrow[8], lrow[8];
    #pragma unroll
    for (int r = 0; r < 8; ++r) { mrow[r] = -1e30f; lrow[r] = 0.f; }

    for (int kv0 = 0; kv0 < SEQ; kv0 += 32) {
        const int cur = (kv0 >> 5) & 1;
        if (wid == 0) {
            if (kv0 + 32 < SEQ) {   // prefetch next chunk into other buffer
                const int nxt = cur ^ 1;
                tdm_load_2d(lds_off(&Ks[nxt][0][0]), Kh + (size_t)(kv0 + 32) * HD,
                            32 * 64, 1, SEQ * HD, 1, 32 * 64);
                tdm_load_2d(lds_off(&Vs[nxt][0][0]), Vt + (kv0 + 32),
                            32, 64, SEQ, HD, SEQ);
                __builtin_amdgcn_s_wait_tensorcnt(2);   // chunk `cur` complete
            } else {
                __builtin_amdgcn_s_wait_tensorcnt(0);
            }
        }
        __syncthreads();   // publish chunk `cur` to all waves

        // S = Q K^T  (16 x 32 per wave; 2 N-tiles, 2 K-steps over hd=64)
        v8f s0 = {}, s1 = {};
        #pragma unroll
        for (int ks = 0; ks < 2; ++ks) {
            v16bf b0 = load_frag(&Ks[cur][0][0], 64, 0,  ks * 32);
            v16bf b1 = load_frag(&Ks[cur][0][0], 64, 16, ks * 32);
            s0 = wmma_bf16(qa[ks], b0, s0);
            s1 = wmma_bf16(qa[ks], b1, s1);
        }

        // online softmax: C-layout row r lives at VGPR slot r within a
        // 16-lane group; reduce across lanes with xor masks 1,2,4,8.
        #pragma unroll
        for (int r = 0; r < 8; ++r) {
            float a0 = s0[r] * scale;
            float a1 = s1[r] * scale;
            float mx = fmaxf(a0, a1);
            #pragma unroll
            for (int m = 1; m < 16; m <<= 1)
                mx = fmaxf(mx, __shfl_xor(mx, m, 32));
            float mnew  = fmaxf(mrow[r], mx);
            float alpha = __expf(mrow[r] - mnew);
            float p0 = __expf(a0 - mnew);
            float p1 = __expf(a1 - mnew);
            float ps = p0 + p1;
            #pragma unroll
            for (int m = 1; m < 16; m <<= 1)
                ps += __shfl_xor(ps, m, 32);
            lrow[r] = lrow[r] * alpha + ps;
            mrow[r] = mnew;
            #pragma unroll
            for (int t = 0; t < 4; ++t) o_acc[t][r] *= alpha;
            // stash P (C-layout -> row-major bf16) for A-fragment reload
            int prow = r + 8 * (lane >> 4);
            int pc   = lane & 15;
            Ps[wid][prow][pc]      = f2bf(p0);
            Ps[wid][prow][pc + 16] = f2bf(p1);
        }
        // intra-wave LDS write->read ordering (CDNA5 split DS counter)
        asm volatile("s_wait_dscnt 0" ::: "memory");

        // O += P @ V   (A = P 16x32, B = V columns contiguous in Vs[d][kv])
        v16bf pa = load_frag(&Ps[wid][0][0], 32, 0, 0);
        #pragma unroll
        for (int t = 0; t < 4; ++t) {
            v16bf bvf = load_frag(&Vs[cur][0][0], 32, t * 16, 0);
            o_acc[t] = wmma_bf16(pa, bvf, o_acc[t]);
        }
        __syncthreads();   // all waves done with buffer `cur` before reuse
    }

    // normalize and write bf16 head-major [B,H,L,64]
    const int grp = lane >> 4;
    const int cn  = lane & 15;
    unsigned short* Oh = O + head_off;
    #pragma unroll
    for (int t = 0; t < 4; ++t)
        #pragma unroll
        for (int r = 0; r < 8; ++r) {
            int l = l0 + wid * 16 + r + 8 * grp;
            int d = t * 16 + cn;
            Oh[(size_t)l * HD + d] = f2bf(o_acc[t][r] / lrow[r]);
        }
}

// ---------- stage 3: output projection ----------
// out[M=4096, N=768] = Ohead(bf16, head-major) @ Wp^T + bp   (f32 output)

__global__ __launch_bounds__(256) void out_proj_kernel(
    const unsigned short* __restrict__ Oi, const float* __restrict__ Wp,
    const float* __restrict__ bp, float* __restrict__ out)
{
    __shared__ unsigned short As[128][40];
    __shared__ unsigned short Bs[128][40];

    const int tid  = threadIdx.x;
    const int wid  = tid >> 5;
    const int lane = tid & 31;
    const int wm = wid & 3;
    const int wn = wid >> 2;
    const int m0 = blockIdx.x * 128;
    const int n0 = blockIdx.y * 128;

    v8f acc[2][4] = {};

    const int lrow = tid >> 3;
    const int lcol = (tid & 7) * 4;

    for (int kt = 0; kt < DIM; kt += 32) {
        const int h  = kt >> 6;              // one head per 32-wide K chunk
        const int d0 = (kt & 63) + lcol;
        #pragma unroll
        for (int p = 0; p < 4; ++p) {
            int r  = p * 32 + lrow;
            int gm = m0 + r;
            int b = gm >> 11, l = gm & 2047;
            const unsigned short* src =
                Oi + (((size_t)b * HEADS + h) * SEQ + l) * HD + d0;
            *(uint2*)&As[r][lcol] = *(const uint2*)src;   // already bf16
            float4 w = *(const float4*)(Wp + (size_t)(n0 + r) * DIM + kt + lcol);
            *(uint2*)&Bs[r][lcol] = make_uint2(pack2(w.x, w.y), pack2(w.z, w.w));
        }
        __syncthreads();
        v16bf af[2], bfr[4];
        #pragma unroll
        for (int i = 0; i < 2; ++i) af[i]  = load_frag(&As[0][0], 40, wm * 32 + i * 16, 0);
        #pragma unroll
        for (int j = 0; j < 4; ++j) bfr[j] = load_frag(&Bs[0][0], 40, wn * 64 + j * 16, 0);
        #pragma unroll
        for (int i = 0; i < 2; ++i)
            #pragma unroll
            for (int j = 0; j < 4; ++j)
                acc[i][j] = wmma_bf16(af[i], bfr[j], acc[i][j]);
        __syncthreads();
    }

    const int grp = lane >> 4;
    const int cn  = lane & 15;
    #pragma unroll
    for (int i = 0; i < 2; ++i)
        #pragma unroll
        for (int j = 0; j < 4; ++j)
            #pragma unroll
            for (int r = 0; r < 8; ++r) {
                int gm = m0 + wm * 32 + i * 16 + r + 8 * grp;
                int gn = n0 + wn * 64 + j * 16 + cn;
                out[(size_t)gm * DIM + gn] = acc[i][j][r] + bp[gn];
            }
}

// ---------- launch ----------

extern "C" void kernel_launch(void* const* d_in, const int* in_sizes, int n_in,
                              void* d_out, int out_size, void* d_ws, size_t ws_size,
                              hipStream_t stream) {
    const float* x  = (const float*)d_in[0];
    const float* y  = (const float*)d_in[1];
    const float* Wq = (const float*)d_in[2];
    const float* bq = (const float*)d_in[3];
    const float* Wk = (const float*)d_in[4];
    const float* bk = (const float*)d_in[5];
    const float* Wv = (const float*)d_in[6];
    const float* bv = (const float*)d_in[7];
    const float* Wp = (const float*)d_in[8];
    const float* bp = (const float*)d_in[9];

    const size_t elems = (size_t)BATCH * HEADS * SEQ * HD;   // 3,145,728
    unsigned short* Qb = (unsigned short*)d_ws;
    unsigned short* Kb = Qb + elems;
    unsigned short* Vb = Kb + elems;   // stored transposed [B,H,64,SEQ]
    unsigned short* Ob = Vb + elems;   // total ws use: 8*elems bytes ~= 24 MB

    dim3 g1(MTOT / 128, DIM / 128, 3);                    // 32 x 6 x 3
    qkv_proj_kernel<<<g1, 256, 0, stream>>>(x, y, Wq, bq, Wk, bk, Wv, bv,
                                            Qb, Kb, Vb);

    dim3 g2(SEQ / 64, BATCH * HEADS);                     // 32 x 24
    attn_kernel<<<g2, 128, 0, stream>>>(Qb, Kb, Vb, Ob);

    dim3 g3(MTOT / 128, DIM / 128);                       // 32 x 6
    out_proj_kernel<<<g3, 256, 0, stream>>>(Ob, Wp, bp, (float*)d_out);
}
